// SelfAttention_39341900431834
// MI455X (gfx1250) — matrix operand order repaired and checked by reading, not compiled, not run on hardware
//
#include <hip/hip_runtime.h>
#include <hip/hip_bf16.h>

// Problem constants (from reference)
#define B_   4
#define C_   2048
#define DIN  1024
#define H_   1024
#define NH   16
#define DH   64

typedef __attribute__((ext_vector_type(16))) _Float16 v16h;
typedef __attribute__((ext_vector_type(8)))  _Float16 v8h;
typedef __attribute__((ext_vector_type(8)))  float    v8f;

static __device__ __forceinline__ v8f wmma_f16(v16h a, v16h b, v8f c) {
  // D = A(16x32 f16) * B(32x16 f16) + C(16x16 f32)
  return __builtin_amdgcn_wmma_f32_16x16x32_f16(false, a, false, b, (short)0, c,
                                                false, false);
}

// A-fragment (16x32, row-major source, leading dim `ld` in elements).
// Lane L<16 holds row L, K in {0..7, 16..23}; lane L>=16 holds row L-16,
// K in {8..15, 24..31}  (per CDNA5 ISA 16-bit A-matrix table).
static __device__ __forceinline__ v16h load_frag_a(const _Float16* base, int ld) {
  const int lane = threadIdx.x & 31;
  const _Float16* r = base + (size_t)(lane & 15) * ld + ((lane >> 4) << 3);
  v8h lo = *reinterpret_cast<const v8h*>(r);
  v8h hi = *reinterpret_cast<const v8h*>(r + 16);
  v16h f;
#pragma unroll
  for (int i = 0; i < 8; ++i) { f[i] = lo[i]; f[i + 8] = hi[i]; }
  return f;
}

// B-fragment (32x16). Source stored "B^T row-major": row n holds the K-dim
// contiguously (leading dim `ld`). Lanes 0-15 hold K in {0..7, 16..23}, lanes
// 16-31 hold K in {8..15, 24..31} (same K order as the A-fragment).
static __device__ __forceinline__ v16h load_frag_b(const _Float16* base, int ld) {
  const int lane = threadIdx.x & 31;
  const _Float16* r = base + (size_t)(lane & 15) * ld + ((lane >> 4) << 3);
  v8h lo = *reinterpret_cast<const v8h*>(r);
  v8h hi = *reinterpret_cast<const v8h*>(r + 16);
  v16h f;
#pragma unroll
  for (int i = 0; i < 8; ++i) { f[i] = lo[i]; f[i + 8] = hi[i]; }
  return f;
}

// ---------------------------------------------------------------------------
// LayerNorm: one row (1024 f32) per 256-thread block, emit f16.
// ---------------------------------------------------------------------------
static __device__ __forceinline__ float block_sum_256(float v, float* sm) {
#pragma unroll
  for (int m = 16; m > 0; m >>= 1) v += __shfl_xor(v, m, 32);
  __syncthreads();
  if ((threadIdx.x & 31) == 0) sm[threadIdx.x >> 5] = v;
  __syncthreads();
  float r = 0.f;
#pragma unroll
  for (int i = 0; i < 8; ++i) r += sm[i];
  return r;
}

__global__ __launch_bounds__(256) void ln_kernel(const float* __restrict__ x,
                                                 const float* __restrict__ g,
                                                 const float* __restrict__ beta,
                                                 _Float16* __restrict__ xn) {
  __shared__ float sm[8];
  const int row = blockIdx.x;
  const float* xr = x + (size_t)row * DIN;
  float v[4];
  float s = 0.f;
#pragma unroll
  for (int i = 0; i < 4; ++i) { v[i] = xr[threadIdx.x + i * 256]; s += v[i]; }
  s = block_sum_256(s, sm);
  const float mu = s * (1.0f / DIN);
  float s2 = 0.f;
#pragma unroll
  for (int i = 0; i < 4; ++i) { float d = v[i] - mu; s2 += d * d; }
  s2 = block_sum_256(s2, sm);
  const float rstd = rsqrtf(s2 * (1.0f / DIN) + 1e-5f);
  _Float16* xo = xn + (size_t)row * DIN;
#pragma unroll
  for (int i = 0; i < 4; ++i) {
    const int c = threadIdx.x + i * 256;
    xo[c] = (_Float16)((v[i] - mu) * rstd * g[c] + beta[c]);
  }
}

// ---------------------------------------------------------------------------
// Transpose + convert 1024x1024 f32 weight [k][n] -> f16 [n][k].
// ---------------------------------------------------------------------------
__global__ __launch_bounds__(256) void transpose_w(const float* __restrict__ in,
                                                   _Float16* __restrict__ out) {
  __shared__ float tile[32][33];
  const int tx = threadIdx.x & 31;
  const int ty = threadIdx.x >> 5;  // 0..7
  const int bx = blockIdx.x * 32;   // n block
  const int by = blockIdx.y * 32;   // k block
#pragma unroll
  for (int r = 0; r < 4; ++r)
    tile[ty + r * 8][tx] = in[(size_t)(by + ty + r * 8) * H_ + bx + tx];
  __syncthreads();
#pragma unroll
  for (int r = 0; r < 4; ++r)
    out[(size_t)(bx + ty + r * 8) * DIN + by + tx] = (_Float16)tile[tx][ty + r * 8];
}

// ---------------------------------------------------------------------------
// WMMA GEMM: C(MxN) = (A(MxK) @ Bt^T + bias) * alpha.
// A row-major f16 (ld=K); Bt is N x K row-major f16 (weights pre-transposed).
// Block: 256 thr (8 waves) -> 64x128 tile; wave -> 32x32; BK=32.
// ---------------------------------------------------------------------------
template <bool OUT_F16>
__global__ __launch_bounds__(256, 2) void gemm_wmma_kernel(
    const _Float16* __restrict__ A,
    const _Float16* __restrict__ Bt,
    const float* __restrict__ bias,
    void* __restrict__ Cout,
    int M, int N, int K, float alpha) {
  __shared__ alignas(32) _Float16 As[64 * 32];
  __shared__ alignas(32) _Float16 Bs[128 * 32];
  const int tid = threadIdx.x;
  const int lane = tid & 31;
  const int w = tid >> 5;
  const int wm = w >> 2, wn = w & 3;
  const int m0 = blockIdx.y * 64;
  const int n0 = blockIdx.x * 128;

  v8f zf = {};
  v8f acc[2][2];
#pragma unroll
  for (int i = 0; i < 2; ++i)
#pragma unroll
    for (int j = 0; j < 2; ++j) acc[i][j] = zf;

  const int arow = tid >> 2, ac8 = (tid & 3) << 3;
  for (int kt = 0; kt < K; kt += 32) {
    __syncthreads();
    *reinterpret_cast<v8h*>(&As[arow * 32 + ac8]) =
        *reinterpret_cast<const v8h*>(&A[(size_t)(m0 + arow) * K + kt + ac8]);
#pragma unroll
    for (int i = 0; i < 2; ++i) {
      const int c = tid + i * 256;
      const int brow = c >> 2, bc8 = (c & 3) << 3;
      *reinterpret_cast<v8h*>(&Bs[brow * 32 + bc8]) =
          *reinterpret_cast<const v8h*>(&Bt[(size_t)(n0 + brow) * K + kt + bc8]);
    }
    if (kt + 32 < K)
      __builtin_prefetch(&A[(size_t)(m0 + arow) * K + kt + 32 + ac8], 0, 0);
    __syncthreads();

    v16h a0 = load_frag_a(&As[(wm * 32 + 0) * 32], 32);
    v16h a1 = load_frag_a(&As[(wm * 32 + 16) * 32], 32);
    v16h b0 = load_frag_b(&Bs[(wn * 32 + 0) * 32], 32);
    v16h b1 = load_frag_b(&Bs[(wn * 32 + 16) * 32], 32);
    acc[0][0] = wmma_f16(a0, b0, acc[0][0]);
    acc[0][1] = wmma_f16(a0, b1, acc[0][1]);
    acc[1][0] = wmma_f16(a1, b0, acc[1][0]);
    acc[1][1] = wmma_f16(a1, b1, acc[1][1]);
  }

  const int half = lane >> 4, li = lane & 15;
#pragma unroll
  for (int fm = 0; fm < 2; ++fm)
#pragma unroll
    for (int fn = 0; fn < 2; ++fn) {
      const int col = n0 + wn * 32 + fn * 16 + li;
      const float bcol = bias[col];
#pragma unroll
      for (int r = 0; r < 8; ++r) {
        const int row = m0 + wm * 32 + fm * 16 + 8 * half + r;
        const float val = (acc[fm][fn][r] + bcol) * alpha;
        if (OUT_F16)
          reinterpret_cast<_Float16*>(Cout)[(size_t)row * N + col] = (_Float16)val;
        else
          reinterpret_cast<float*>(Cout)[(size_t)row * N + col] = val;
      }
    }
}

// ---------------------------------------------------------------------------
// Flash attention: block = 128 thr (4 waves) owns one (b, h, 64-row Q tile).
// Wave w owns 16 query rows. Q pre-scaled by 1/sqrt(Dh) in the Q GEMM.
// __launch_bounds__(128, 1): full VGPR budget -> no scratch spills in loop.
// ---------------------------------------------------------------------------
__global__ __launch_bounds__(128, 1) void attn_kernel(
    const _Float16* __restrict__ Q,
    const _Float16* __restrict__ K,
    const _Float16* __restrict__ V,
    const unsigned char* __restrict__ amask,
    _Float16* __restrict__ AV) {
  __shared__ alignas(32) _Float16 Ks[64 * 64];
  __shared__ alignas(32) _Float16 Vt[64 * 64];   // V transposed: [d][key]
  __shared__ alignas(32) _Float16 Ps[4 * 16 * 64];
  const int tid = threadIdx.x;
  const int lane = tid & 31;
  const int w = tid >> 5;
  const int half = lane >> 4, li = lane & 15;
  const int qt = blockIdx.x, h = blockIdx.y, b = blockIdx.z;
  const int q0 = qt * 64;

  const size_t qbase = ((size_t)b * C_ + q0 + w * 16) * H_ + (size_t)h * DH;
  v16h qa0 = load_frag_a(Q + qbase, H_);        // d 0..31
  v16h qa1 = load_frag_a(Q + qbase + 32, H_);   // d 32..63

  v8f zf = {};
  v8f o[4];
#pragma unroll
  for (int nt = 0; nt < 4; ++nt) o[nt] = zf;
  float mrow[8], lrow[8];
#pragma unroll
  for (int r = 0; r < 8; ++r) { mrow[r] = -1e30f; lrow[r] = 0.f; }

  _Float16* Pw = &Ps[w * 16 * 64];

  for (int jt = 0; jt <= qt; ++jt) {
    const int j0 = jt * 64;
    __syncthreads();  // protect LDS tiles from previous iteration's readers
    // Stage K tile (row-major) and V tile (transposed) into LDS.
#pragma unroll
    for (int i = 0; i < 4; ++i) {
      const int c = tid + i * 128;
      const int row = c >> 3, c8 = (c & 7) << 3;
      const size_t gofs = ((size_t)b * C_ + j0 + row) * H_ + (size_t)h * DH + c8;
      *reinterpret_cast<v8h*>(&Ks[row * 64 + c8]) =
          *reinterpret_cast<const v8h*>(&K[gofs]);
      v8h vv = *reinterpret_cast<const v8h*>(&V[gofs]);
#pragma unroll
      for (int e = 0; e < 8; ++e) Vt[(c8 + e) * 64 + row] = vv[e];
    }
    __syncthreads();

    // Scores: S(16x64) = Q(16x64) @ K^T  (Q already scaled by 1/8).
    v8f s[4];
#pragma unroll
    for (int nt = 0; nt < 4; ++nt) {
      v16h kb0 = load_frag_b(&Ks[nt * 16 * 64], 64);       // keys, d 0..31
      v16h kb1 = load_frag_b(&Ks[nt * 16 * 64 + 32], 64);  // keys, d 32..63
      v8f sa = zf;
      sa = wmma_f16(qa0, kb0, sa);
      sa = wmma_f16(qa1, kb1, sa);
      s[nt] = sa;
    }

    // Causal + attention mask (C-layout: row = 8*half + r, col = li).
#pragma unroll
    for (int nt = 0; nt < 4; ++nt) {
      const int key = j0 + nt * 16 + li;
      const bool am = amask[(size_t)b * C_ + key] != 0;
#pragma unroll
      for (int r = 0; r < 8; ++r) {
        const int qq = q0 + w * 16 + 8 * half + r;
        if (key > qq || !am) s[nt][r] = -1e30f;
      }
    }

    // Online softmax update (reduce across the 16 lanes of each half).
#pragma unroll
    for (int r = 0; r < 8; ++r) {
      float t = fmaxf(fmaxf(s[0][r], s[1][r]), fmaxf(s[2][r], s[3][r]));
#pragma unroll
      for (int m = 8; m > 0; m >>= 1) t = fmaxf(t, __shfl_xor(t, m, 32));
      const float mnew = fmaxf(mrow[r], t);
      const float sc = __expf(mrow[r] - mnew);
      mrow[r] = mnew;
      float rs = 0.f;
#pragma unroll
      for (int nt = 0; nt < 4; ++nt) {
        const float p = __expf(s[nt][r] - mnew);
        s[nt][r] = p;
        rs += p;
      }
#pragma unroll
      for (int m = 8; m > 0; m >>= 1) rs += __shfl_xor(rs, m, 32);
      lrow[r] = lrow[r] * sc + rs;
#pragma unroll
      for (int nt = 0; nt < 4; ++nt) o[nt][r] *= sc;
    }

    // Re-layout probabilities through LDS into A-fragments.
#pragma unroll
    for (int nt = 0; nt < 4; ++nt)
#pragma unroll
      for (int r = 0; r < 8; ++r)
        Pw[(8 * half + r) * 64 + nt * 16 + li] = (_Float16)s[nt][r];
    __syncthreads();

    v16h pa0 = load_frag_a(Pw, 64);        // keys 0..31
    v16h pa1 = load_frag_a(Pw + 32, 64);   // keys 32..63
#pragma unroll
    for (int nt = 0; nt < 4; ++nt) {
      v16h vb0 = load_frag_b(&Vt[nt * 16 * 64], 64);       // B[key][d], keys 0..31
      v16h vb1 = load_frag_b(&Vt[nt * 16 * 64 + 32], 64);  // keys 32..63
      o[nt] = wmma_f16(pa0, vb0, o[nt]);
      o[nt] = wmma_f16(pa1, vb1, o[nt]);
    }
  }

  // Normalize and store AV (f16) in [b][c][h*64+d] layout.
#pragma unroll
  for (int r = 0; r < 8; ++r) {
    const float inv = 1.0f / lrow[r];
    const int row = q0 + w * 16 + 8 * half + r;
#pragma unroll
    for (int nt = 0; nt < 4; ++nt)
      AV[((size_t)b * C_ + row) * H_ + (size_t)h * DH + nt * 16 + li] =
          (_Float16)(o[nt][r] * inv);
  }
}

// ---------------------------------------------------------------------------
extern "C" void kernel_launch(void* const* d_in, const int* in_sizes, int n_in,
                              void* d_out, int out_size, void* d_ws, size_t ws_size,
                              hipStream_t stream) {
  (void)in_sizes; (void)n_in; (void)out_size; (void)ws_size;
  const float*         x     = (const float*)d_in[0];
  const unsigned char* amask = (const unsigned char*)d_in[1];
  const float* ln_g = (const float*)d_in[2];
  const float* ln_b = (const float*)d_in[3];
  const float* wq = (const float*)d_in[4];
  const float* bq = (const float*)d_in[5];
  const float* wk = (const float*)d_in[6];
  const float* bk = (const float*)d_in[7];
  const float* wv = (const float*)d_in[8];
  const float* bv = (const float*)d_in[9];
  const float* wo = (const float*)d_in[10];
  const float* bo = (const float*)d_in[11];
  float* out = (float*)d_out;

  char* ws = (char*)d_ws;
  const size_t ROWS = (size_t)B_ * C_;                      // 8192
  const size_t XN_B = ROWS * DIN * sizeof(_Float16);        // 16 MB
  const size_t W_B  = (size_t)DIN * H_ * sizeof(_Float16);  // 2 MB
  size_t off = 0;
  _Float16* xn  = (_Float16*)(ws + off); off += XN_B;
  _Float16* wqT = (_Float16*)(ws + off); off += W_B;
  _Float16* wkT = (_Float16*)(ws + off); off += W_B;
  _Float16* wvT = (_Float16*)(ws + off); off += W_B;
  _Float16* woT = (_Float16*)(ws + off); off += W_B;
  _Float16* Qb  = (_Float16*)(ws + off); off += XN_B;
  _Float16* Kb  = (_Float16*)(ws + off); off += XN_B;
  _Float16* Vb  = (_Float16*)(ws + off); off += XN_B;
  _Float16* AVb = (_Float16*)(ws + off); off += XN_B;

  // 1. LayerNorm -> f16
  ln_kernel<<<dim3((unsigned)ROWS), dim3(256), 0, stream>>>(x, ln_g, ln_b, xn);

  // 2. Weight transpose + f32 -> f16
  dim3 tg(H_ / 32, DIN / 32);
  transpose_w<<<tg, 256, 0, stream>>>(wq, wqT);
  transpose_w<<<tg, 256, 0, stream>>>(wk, wkT);
  transpose_w<<<tg, 256, 0, stream>>>(wv, wvT);
  transpose_w<<<tg, 256, 0, stream>>>(wo, woT);

  // 3. QKV projections (Q absorbs the 1/sqrt(Dh) score scale)
  dim3 gg(H_ / 128, (unsigned)(ROWS / 64));
  gemm_wmma_kernel<true><<<gg, 256, 0, stream>>>(xn, wqT, bq, Qb,
                                                 (int)ROWS, H_, DIN, 0.125f);
  gemm_wmma_kernel<true><<<gg, 256, 0, stream>>>(xn, wkT, bk, Kb,
                                                 (int)ROWS, H_, DIN, 1.0f);
  gemm_wmma_kernel<true><<<gg, 256, 0, stream>>>(xn, wvT, bv, Vb,
                                                 (int)ROWS, H_, DIN, 1.0f);

  // 4. Causal flash attention
  attn_kernel<<<dim3(C_ / 64, NH, B_), 128, 0, stream>>>(Qb, Kb, Vb, amask, AVb);

  // 5. Output projection -> f32 d_out
  gemm_wmma_kernel<false><<<gg, 256, 0, stream>>>(AVb, woT, bo, (void*)out,
                                                  (int)ROWS, H_, DIN, 1.0f);
}
